// Cross_Attention_20993800143438
// MI455X (gfx1250) — compile-verified, matrix operand order checked by reading, not run on hardware
//
#include <hip/hip_runtime.h>
#include <cstddef>
#include <cstdint>

// ---------------------------------------------------------------------------
// types for WMMA / TDM
// ---------------------------------------------------------------------------
typedef __bf16        bf16x16 __attribute__((ext_vector_type(16)));
typedef float         f32x8   __attribute__((ext_vector_type(8)));
typedef unsigned int  u32x8   __attribute__((ext_vector_type(8)));
typedef unsigned int  u32x4   __attribute__((ext_vector_type(4)));
typedef int           i32x8   __attribute__((ext_vector_type(8)));
typedef int           i32x4   __attribute__((ext_vector_type(4)));

#define DEVINL __device__ __forceinline__

#if __has_builtin(__builtin_amdgcn_tensor_load_to_lds) && \
    __has_builtin(__builtin_amdgcn_s_wait_tensorcnt)
#define USE_TDM 1
#else
#define USE_TDM 0
#endif

DEVINL unsigned short f2bf(float f) {
    unsigned int u = __builtin_bit_cast(unsigned int, f);
    unsigned int r = u + 0x7FFFu + ((u >> 16) & 1u);
    return (unsigned short)(r >> 16);
}

DEVINL f32x8 zero8() {
    f32x8 z = {0.f, 0.f, 0.f, 0.f, 0.f, 0.f, 0.f, 0.f};
    return z;
}

DEVINL f32x8 wmma_bf16(u32x8 a, u32x8 b, f32x8 c) {
    bf16x16 av = __builtin_bit_cast(bf16x16, a);
    bf16x16 bv = __builtin_bit_cast(bf16x16, b);
    return __builtin_amdgcn_wmma_f32_16x16x32_bf16(false, av, false, bv,
                                                   (short)0, c, false, false);
}

// A/B fragment from a row-major bf16(u16) array (A: row = M-row of lane;
// B stored transposed [n][k]: row = N-column of lane). kg = lane>>4.
DEVINL u32x8 frag_row(const unsigned short* base, int ld, int row, int k0, int kg) {
    const unsigned short* p = base + (size_t)row * ld + k0;
    u32x4 lo = *(const u32x4*)(p + kg * 8);
    u32x4 hi = *(const u32x4*)(p + 16 + kg * 8);
    u32x8 r;
    r[0] = lo[0]; r[1] = lo[1]; r[2] = lo[2]; r[3] = lo[3];
    r[4] = hi[0]; r[5] = hi[1]; r[6] = hi[2]; r[7] = hi[3];
    return r;
}

// Strided fragment: element j at base[(kg*8+j)*stride], hi half +16.
DEVINL u32x8 frag_strided(const unsigned short* base, size_t stride, int kg) {
    u32x8 r;
#pragma unroll
    for (int i = 0; i < 4; ++i) {
        unsigned int a0 = base[(size_t)(kg * 8 + 2 * i) * stride];
        unsigned int a1 = base[(size_t)(kg * 8 + 2 * i + 1) * stride];
        r[i] = a0 | (a1 << 16);
    }
#pragma unroll
    for (int i = 0; i < 4; ++i) {
        unsigned int a0 = base[(size_t)(16 + kg * 8 + 2 * i) * stride];
        unsigned int a1 = base[(size_t)(16 + kg * 8 + 2 * i + 1) * stride];
        r[4 + i] = a0 | (a1 << 16);
    }
    return r;
}

#if USE_TDM
// Issue a TDM 2D tile load (bf16): tile_y rows x tile_x elems, row stride
// stride_x elems, from gptr into LDS at lds_addr. D# packed per CDNA5 ISA
// 8.3/8.4 (count=1, data_size=1(2B), type=2). 6-arg clang-23 builtin form.
DEVINL void tdm_load_2d(const unsigned short* gptr, unsigned lds_addr,
                        unsigned tile_x, unsigned tile_y,
                        unsigned tensor_x, unsigned tensor_y,
                        unsigned stride_x) {
    unsigned long long ga = (unsigned long long)(uintptr_t)gptr;
    u32x4 g0;
    g0[0] = 1u;                                       // count=1 (valid)
    g0[1] = lds_addr;                                 // lds_addr[31:0]
    g0[2] = (unsigned)(ga & 0xffffffffu);             // global_addr[31:0]
    g0[3] = (unsigned)((ga >> 32) & 0x01ffffffu) | (2u << 30);  // addr hi | type=2
    i32x8 g1;
    g1[0] = (int)(1u << 16);                          // data_size=1 -> 2 bytes
    g1[1] = (int)((tensor_x & 0xffffu) << 16);        // tensor_dim0[15:0]
    g1[2] = (int)(((tensor_x >> 16) & 0xffffu) |      // tensor_dim0[31:16]
                  ((tensor_y & 0xffffu) << 16));      // tensor_dim1[15:0]
    g1[3] = (int)(((tensor_y >> 16) & 0xffffu) |      // tensor_dim1[31:16]
                  ((tile_x & 0xffffu) << 16));        // tile_dim0
    g1[4] = (int)(tile_y & 0xffffu);                  // tile_dim1 (tile_dim2=0)
    g1[5] = (int)stride_x;                            // tensor_dim0_stride[31:0]
    g1[6] = 0;                                        // stride hi / dim1_stride lo
    g1[7] = 0;
    i32x4 z4 = {0, 0, 0, 0};
    i32x8 z8 = {0, 0, 0, 0, 0, 0, 0, 0};
    __builtin_amdgcn_tensor_load_to_lds(g0, g1, z4, z4, z8, 0);
}
#endif

// ---------------------------------------------------------------------------
// problem constants
// ---------------------------------------------------------------------------
#define BB     2
#define NTOK   2048
#define DIM    1024
#define SDIM   512
#define QKV3   3072
#define NH     16
#define HD     64
#define BN     (BB * NTOK)           // 4096
#define CH_SCALE   (1.0f / 64.0f)
#define TOK_SCALE  0.02209708691207961f   // 1/sqrt(2048)

// ---------------------------------------------------------------------------
// elementwise / transpose kernels
// ---------------------------------------------------------------------------
__global__ void ca_cvt_bf16(const float* __restrict__ in,
                            unsigned short* __restrict__ out, size_t total) {
    size_t i = (size_t)blockIdx.x * blockDim.x + threadIdx.x;
    if (i < total) out[i] = f2bf(in[i]);
}

// W[K][N] f32 -> Wt[N][K] bf16, 32x32 LDS tiles.
__global__ __launch_bounds__(256) void ca_transpose_cvt(
        const float* __restrict__ W, unsigned short* __restrict__ Wt,
        int K, int N) {
    __shared__ float tile[32][33];
    const int nb = blockIdx.x * 32;   // n base
    const int kb = blockIdx.y * 32;   // k base
    const int tx = threadIdx.x & 31, ty = threadIdx.x >> 5;  // ty 0..7
#pragma unroll
    for (int i = 0; i < 32; i += 8)
        tile[ty + i][tx] = W[(size_t)(kb + ty + i) * N + nb + tx];
    __syncthreads();
#pragma unroll
    for (int i = 0; i < 32; i += 8)
        Wt[(size_t)(nb + ty + i) * K + kb + tx] = f2bf(tile[tx][ty + i]);
}

__global__ void ca_copy_v(const float* __restrict__ qkv,
                          float* __restrict__ out0, int total) {
    int i = blockIdx.x * blockDim.x + threadIdx.x;
    if (i < total) {
        int bn = i >> 10;
        int c  = i & 1023;
        out0[i] = qkv[(size_t)bn * QKV3 + 2048 + c];
    }
}

__global__ void ca_scale_co(const float* __restrict__ co,
                            const float* __restrict__ colsum,
                            unsigned short* __restrict__ cob, int total) {
    int i = blockIdx.x * blockDim.x + threadIdx.x;
    if (i < total) {
        int m  = (i >> 6) & 2047;
        int bh = i >> 17;
        cob[i] = f2bf(co[i] / colsum[(size_t)bh * NTOK + m]);
    }
}

// ---------------------------------------------------------------------------
// K1: C[MxN] f32 = A[MxK] bf16 @ Bt[NxK]^T bf16.  128x128 tile, 8 waves.
// TDM double-buffered staging of A/B tiles into LDS.
// ---------------------------------------------------------------------------
__global__ __launch_bounds__(256) void ca_gemm_bf16(
        const unsigned short* __restrict__ A,
        const unsigned short* __restrict__ Bt,
        float* __restrict__ C, int M, int N, int K) {
    __shared__ unsigned short Ald[2][128 * 32];   // [m][k]
    __shared__ unsigned short Bld[2][128 * 32];   // [n][k]
    const int tid  = threadIdx.x;
    const int wave = tid >> 5, lane = tid & 31;
    const int lm = lane & 15, kg = lane >> 4, lh = kg;
    const int n0 = blockIdx.x * 128, m0 = blockIdx.y * 128;
    const int wr = wave & 1, wc = wave >> 1;      // wave: 64 rows x 32 cols
    const int nchunks = K >> 5;
    const int wave0 = (__builtin_amdgcn_readfirstlane(wave) == 0);

    f32x8 acc[4][2];
#pragma unroll
    for (int i = 0; i < 4; ++i)
#pragma unroll
        for (int j = 0; j < 2; ++j) acc[i][j] = zero8();

#if USE_TDM
    if (wave0) {   // prologue: chunk 0 -> buffer 0
        tdm_load_2d(A  + (size_t)m0 * K, (unsigned)(uintptr_t)&Ald[0][0],
                    32, 128, (unsigned)K, (unsigned)M, (unsigned)K);
        tdm_load_2d(Bt + (size_t)n0 * K, (unsigned)(uintptr_t)&Bld[0][0],
                    32, 128, (unsigned)K, (unsigned)N, (unsigned)K);
    }
#endif
    for (int t = 0; t < nchunks; ++t) {
        const int cur = t & 1;
#if USE_TDM
        if (wave0) {
            if (t + 1 < nchunks) {
                int kc = (t + 1) << 5;
                tdm_load_2d(A + (size_t)m0 * K + kc,
                            (unsigned)(uintptr_t)&Ald[cur ^ 1][0],
                            32, 128, (unsigned)K, (unsigned)M, (unsigned)K);
                tdm_load_2d(Bt + (size_t)n0 * K + kc,
                            (unsigned)(uintptr_t)&Bld[cur ^ 1][0],
                            32, 128, (unsigned)K, (unsigned)N, (unsigned)K);
                __builtin_amdgcn_s_wait_tensorcnt(2);  // chunk t complete
            } else {
                __builtin_amdgcn_s_wait_tensorcnt(0);
            }
        }
#else
        {   // manual staging: thread -> (row = t>>1, half = t&1), contiguous
            int kc  = t << 5;
            int row = tid >> 1, half = tid & 1;
            *(u32x4*)(&Ald[cur][row * 32 + half * 16]) =
                *(const u32x4*)(A + (size_t)(m0 + row) * K + kc + half * 16);
            *(u32x4*)(&Bld[cur][row * 32 + half * 16]) =
                *(const u32x4*)(Bt + (size_t)(n0 + row) * K + kc + half * 16);
        }
#endif
        __syncthreads();

        u32x8 bfr[2];
#pragma unroll
        for (int j = 0; j < 2; ++j)
            bfr[j] = frag_row(&Bld[cur][0], 32, wc * 32 + j * 16 + lm, 0, kg);
#pragma unroll
        for (int i = 0; i < 4; ++i) {
            u32x8 af = frag_row(&Ald[cur][0], 32, wr * 64 + i * 16 + lm, 0, kg);
#pragma unroll
            for (int j = 0; j < 2; ++j)
                acc[i][j] = wmma_bf16(af, bfr[j], acc[i][j]);
        }
        __syncthreads();
    }

    // epilogue: pointer-increment stores (no per-element 64-bit muls)
#pragma unroll
    for (int i = 0; i < 4; ++i)
#pragma unroll
        for (int j = 0; j < 2; ++j) {
            float* p = C + (size_t)(m0 + wr * 64 + i * 16 + lh * 8) * (size_t)N
                         + (n0 + wc * 32 + j * 16 + lm);
#pragma unroll
            for (int r = 0; r < 8; ++r) { *p = acc[i][j][r]; p += N; }
        }
}

// ---------------------------------------------------------------------------
// K2: channel path per (b,h): chan_dots -> softmax -> chan_out (f32 out)
// ---------------------------------------------------------------------------
__global__ __launch_bounds__(256) void ca_chan(
        const unsigned short* __restrict__ qkvb,
        const unsigned short* __restrict__ qkvsb,
        float* __restrict__ chan_out) {
    __shared__ float sdots[64 * 64];            // [d][e]
    __shared__ unsigned short sattn[64 * 64];   // transposed [e][d] bf16
    const int bh = blockIdx.x, b = bh >> 4, h = bh & 15;
    const int tid  = threadIdx.x;
    const int wave = tid >> 5, lane = tid & 31;
    const int lm = lane & 15, kg = lane >> 4, lh = kg;

    const unsigned short* qbase  = qkvb  + (size_t)b * NTOK * QKV3 + h * HD;
    const unsigned short* ksbase = qkvsb + (size_t)b * NTOK * QKV3 + 1024 + h * HD;
    const unsigned short* vbase  = qkvb  + (size_t)b * NTOK * QKV3 + 2048 + h * HD;

    // ---- phase 1: chan_dots[d,e] = sum_n q[n,d]*ks[n,e], K=2048 -------------
    const int dr = wave & 3;
    const int ep = wave >> 2;
    f32x8 cd[2] = {zero8(), zero8()};
    for (int nc = 0; nc < NTOK; nc += 32) {
        u32x8 af = frag_strided(qbase + (size_t)nc * QKV3 + dr * 16 + lm,
                                QKV3, kg);
#pragma unroll
        for (int j = 0; j < 2; ++j) {
            int e = (ep * 2 + j) * 16 + lm;
            u32x8 bf = frag_strided(ksbase + (size_t)nc * QKV3 + e, QKV3, kg);
            cd[j] = wmma_bf16(af, bf, cd[j]);
        }
    }
#pragma unroll
    for (int j = 0; j < 2; ++j)
#pragma unroll
        for (int r = 0; r < 8; ++r) {
            int d = dr * 16 + lh * 8 + r;
            int e = (ep * 2 + j) * 16 + lm;
            sdots[d * 64 + e] = cd[j][r] * CH_SCALE;
        }
    __syncthreads();

    // ---- softmax over e per row d, write transposed bf16 --------------------
    if (tid < 64) {
        float mx = -1e30f;
        for (int e = 0; e < 64; ++e) mx = fmaxf(mx, sdots[tid * 64 + e]);
        float s = 0.f;
        for (int e = 0; e < 64; ++e) s += __expf(sdots[tid * 64 + e] - mx);
        float inv = 1.f / s;
        for (int e = 0; e < 64; ++e)
            sattn[e * 64 + tid] = f2bf(__expf(sdots[tid * 64 + e] - mx) * inv);
    }
    __syncthreads();

    // ---- phase 2: chan_out[m,e] = sum_d v[m,d]*attn[d,e], K=64 --------------
    u32x8 bfr[4][2];
#pragma unroll
    for (int et = 0; et < 4; ++et)
#pragma unroll
        for (int kc = 0; kc < 2; ++kc)
            bfr[et][kc] = frag_row(sattn, 64, et * 16 + lm, kc * 32, kg);

    for (int mc = 0; mc < NTOK; mc += 128) {
        int mrow = mc + wave * 16;
        f32x8 oc[4] = {zero8(), zero8(), zero8(), zero8()};
#pragma unroll
        for (int kc = 0; kc < 2; ++kc) {
            u32x8 af = frag_row(vbase, QKV3, mrow + lm, kc * 32, kg);
#pragma unroll
            for (int et = 0; et < 4; ++et)
                oc[et] = wmma_bf16(af, bfr[et][kc], oc[et]);
        }
#pragma unroll
        for (int et = 0; et < 4; ++et) {
            float* p = chan_out + ((size_t)bh * NTOK + mrow + lh * 8) * HD
                                + et * 16 + lm;
#pragma unroll
            for (int r = 0; r < 8; ++r) { *p = oc[et][r]; p += HD; }
        }
    }
}

// ---------------------------------------------------------------------------
// K3: column stats of S = qs @ k^T * ts : colmax[m], colsum[m] (softmax axis=n)
// grid: (32 bh, 16 m-chunks of 128).  wave w owns 16 columns.
// ---------------------------------------------------------------------------
__global__ __launch_bounds__(256) void ca_colstats(
        const unsigned short* __restrict__ qkvb,
        const unsigned short* __restrict__ qkvsb,
        float* __restrict__ colmax, float* __restrict__ colsum) {
    __shared__ float redM[256], redS[256];
    const int bh = blockIdx.x, b = bh >> 4, h = bh & 15;
    const int m0 = blockIdx.y * 128;
    const int tid  = threadIdx.x;
    const int wave = tid >> 5, lane = tid & 31;
    const int lm = lane & 15, kg = lane >> 4;

    const unsigned short* qsbase = qkvsb + (size_t)b * NTOK * QKV3 + h * HD;
    const unsigned short* kbase  = qkvb  + (size_t)b * NTOK * QKV3 + 1024 + h * HD;

    const int mcol = m0 + wave * 16 + lm;
    u32x8 bf0 = frag_row(kbase, QKV3, mcol, 0, kg);
    u32x8 bf1 = frag_row(kbase, QKV3, mcol, 32, kg);

    float runM = -1e30f, runS = 0.f;
    for (int nc = 0; nc < NTOK; nc += 16) {
        f32x8 c = zero8();
        c = wmma_bf16(frag_row(qsbase, QKV3, nc + lm, 0, kg), bf0, c);
        c = wmma_bf16(frag_row(qsbase, QKV3, nc + lm, 32, kg), bf1, c);
        float tm = -1e30f;
#pragma unroll
        for (int r = 0; r < 8; ++r) tm = fmaxf(tm, c[r] * TOK_SCALE);
        float nM = fmaxf(runM, tm);
        float add = 0.f;
#pragma unroll
        for (int r = 0; r < 8; ++r) add += __expf(c[r] * TOK_SCALE - nM);
        runS = runS * __expf(runM - nM) + add;
        runM = nM;
    }
    redM[tid] = runM; redS[tid] = runS;
    __syncthreads();
    if (lane < 16) {
        float oM = redM[tid + 16], oS = redS[tid + 16];
        float fM = fmaxf(runM, oM);
        float fS = runS * __expf(runM - fM) + oS * __expf(oM - fM);
        colmax[(size_t)bh * NTOK + mcol] = fM;
        colsum[(size_t)bh * NTOK + mcol] = fS;
    }
}

// ---------------------------------------------------------------------------
// K5: out[n,e] = sum_m exp(s[n,m]-colmax[m]) * chan_out_scaled[m,e]
// grid: (32 bh, 16 n-chunks of 128). wave w owns n rows [w*16, w*16+16).
// ---------------------------------------------------------------------------
__global__ __launch_bounds__(256) void ca_passb(
        const unsigned short* __restrict__ qkvb,
        const unsigned short* __restrict__ qkvsb,
        const unsigned short* __restrict__ cob,
        const float* __restrict__ colmax,
        float* __restrict__ out1) {
    __shared__ unsigned short Pld[128 * 128];   // [n][m] bf16
    __shared__ unsigned short COld[64 * 128];   // transposed [e][m] bf16
    const int bh = blockIdx.x, b = bh >> 4, h = bh & 15;
    const int n0 = blockIdx.y * 128;
    const int tid  = threadIdx.x;
    const int wave = tid >> 5, lane = tid & 31;
    const int lm = lane & 15, kg = lane >> 4, lh = kg;

    const unsigned short* qsbase = qkvsb + (size_t)b * NTOK * QKV3 + h * HD;
    const unsigned short* kbase  = qkvb  + (size_t)b * NTOK * QKV3 + 1024 + h * HD;
    const unsigned short* cobase = cob + (size_t)bh * NTOK * HD;
    const float*          cmbase = colmax + (size_t)bh * NTOK;

    const int nrow = n0 + wave * 16;
    u32x8 qa0 = frag_row(qsbase, QKV3, nrow + lm, 0, kg);
    u32x8 qa1 = frag_row(qsbase, QKV3, nrow + lm, 32, kg);

    f32x8 acc[4] = {zero8(), zero8(), zero8(), zero8()};

    for (int mc = 0; mc < NTOK; mc += 128) {
        {   // stage chan_out_scaled chunk transposed: thread -> (e, 32 m's)
            int e = tid >> 2, mg = tid & 3;
            for (int j = 0; j < 32; ++j)
                COld[e * 128 + mg * 32 + j] =
                    cobase[(size_t)(mc + mg * 32 + j) * HD + e];
        }
        // compute S tiles, form P = exp(s*ts - colmax[m]), write to LDS
#pragma unroll
        for (int mt = 0; mt < 8; ++mt) {
            int mcol = mc + mt * 16 + lm;
            u32x8 kb0 = frag_row(kbase, QKV3, mcol, 0, kg);
            u32x8 kb1 = frag_row(kbase, QKV3, mcol, 32, kg);
            f32x8 s = zero8();
            s = wmma_bf16(qa0, kb0, s);
            s = wmma_bf16(qa1, kb1, s);
            float cm = cmbase[mcol];
#pragma unroll
            for (int r = 0; r < 8; ++r) {
                int nn = wave * 16 + lh * 8 + r;
                Pld[nn * 128 + mt * 16 + lm] =
                    f2bf(__expf(s[r] * TOK_SCALE - cm));
            }
        }
        __syncthreads();

        // acc += P[16n x 128m] @ CO[128m x 64e]
#pragma unroll
        for (int kc = 0; kc < 128; kc += 32) {
            u32x8 pa = frag_row(Pld, 128, wave * 16 + lm, kc, kg);
#pragma unroll
            for (int et = 0; et < 4; ++et) {
                u32x8 cb = frag_row(COld, 128, et * 16 + lm, kc, kg);
                acc[et] = wmma_bf16(pa, cb, acc[et]);
            }
        }
        __syncthreads();
    }

    // write merged-head layout: out1[(b*2048+n)*1024 + h*64 + e]
#pragma unroll
    for (int et = 0; et < 4; ++et) {
        float* p = out1 + ((size_t)b * NTOK + nrow + lh * 8) * DIM
                        + h * HD + et * 16 + lm;
#pragma unroll
        for (int r = 0; r < 8; ++r) { *p = acc[et][r]; p += DIM; }
    }
}

// ---------------------------------------------------------------------------
// host: orchestrate
// ---------------------------------------------------------------------------
extern "C" void kernel_launch(void* const* d_in, const int* in_sizes, int n_in,
                              void* d_out, int out_size, void* d_ws, size_t ws_size,
                              hipStream_t stream) {
    (void)in_sizes; (void)n_in; (void)out_size; (void)ws_size;
    const float* x   = (const float*)d_in[0];
    const float* y   = (const float*)d_in[1];
    const float* W   = (const float*)d_in[2];
    const float* Ws  = (const float*)d_in[3];
    float* out0 = (float*)d_out;                 // merge_heads(v) = v slice
    float* out1 = (float*)d_out + (size_t)BN * DIM;

    char* ws = (char*)d_ws;
    size_t off = 0;
    auto alloc = [&](size_t bytes) {
        size_t o = off;
        off = (off + bytes + 255) & ~(size_t)255;
        return o;
    };
    const size_t QKV_ELEMS = (size_t)BN * QKV3;
    float*          qkv    = (float*)(ws + alloc(QKV_ELEMS * 4));
    float*          qkvs   = (float*)(ws + alloc(QKV_ELEMS * 4));
    unsigned short* xb     = (unsigned short*)(ws + alloc((size_t)BN * DIM * 2));
    unsigned short* yb     = (unsigned short*)(ws + alloc((size_t)BN * SDIM * 2));
    unsigned short* Wtb    = (unsigned short*)(ws + alloc((size_t)DIM * QKV3 * 2));
    unsigned short* Wstb   = (unsigned short*)(ws + alloc((size_t)SDIM * QKV3 * 2));
    unsigned short* qkvb   = (unsigned short*)(ws + alloc(QKV_ELEMS * 2));
    unsigned short* qkvsb  = (unsigned short*)(ws + alloc(QKV_ELEMS * 2));
    float*          co     = (float*)(ws + alloc((size_t)BB * NH * NTOK * HD * 4));
    unsigned short* cob    = (unsigned short*)(ws + alloc((size_t)BB * NH * NTOK * HD * 2));
    float*          colmax = (float*)(ws + alloc((size_t)BB * NH * NTOK * 4));
    float*          colsum = (float*)(ws + alloc((size_t)BB * NH * NTOK * 4));

    auto blocks = [](size_t total) { return (unsigned)((total + 255) / 256); };

    // 1. convert activations, transpose+convert weights (Wt[N][K] bf16)
    ca_cvt_bf16<<<blocks((size_t)BN * DIM), 256, 0, stream>>>(x, xb, (size_t)BN * DIM);
    ca_cvt_bf16<<<blocks((size_t)BN * SDIM), 256, 0, stream>>>(y, yb, (size_t)BN * SDIM);
    ca_transpose_cvt<<<dim3(QKV3 / 32, DIM / 32), 256, 0, stream>>>(W, Wtb, DIM, QKV3);
    ca_transpose_cvt<<<dim3(QKV3 / 32, SDIM / 32), 256, 0, stream>>>(Ws, Wstb, SDIM, QKV3);

    // 2. projections (TDM-staged, double-buffered bf16 WMMA GEMM)
    ca_gemm_bf16<<<dim3(QKV3 / 128, BN / 128), 256, 0, stream>>>(xb, Wtb, qkv, BN, QKV3, DIM);
    ca_gemm_bf16<<<dim3(QKV3 / 128, BN / 128), 256, 0, stream>>>(yb, Wstb, qkvs, BN, QKV3, SDIM);

    // 3. bf16 copies of projections for attention WMMAs
    ca_cvt_bf16<<<blocks(QKV_ELEMS), 256, 0, stream>>>(qkv, qkvb, QKV_ELEMS);
    ca_cvt_bf16<<<blocks(QKV_ELEMS), 256, 0, stream>>>(qkvs, qkvsb, QKV_ELEMS);

    // 4. output 0 = v (merge(split(v)) is identity)
    ca_copy_v<<<blocks((size_t)BN * DIM), 256, 0, stream>>>(qkv, out0, BN * DIM);

    // 5. channel attention -> chan_out
    ca_chan<<<BB * NH, 256, 0, stream>>>(qkvb, qkvsb, co);

    // 6. column softmax statistics for token attention (axis = n)
    ca_colstats<<<dim3(BB * NH, NTOK / 128), 256, 0, stream>>>(qkvb, qkvsb, colmax, colsum);

    // 7. fold 1/colsum into chan_out rows, convert to bf16
    ca_scale_co<<<blocks((size_t)BB * NH * NTOK * HD), 256, 0, stream>>>(
        co, colsum, cob, BB * NH * NTOK * HD);

    // 8. pass B: out = exp(S - colmax) @ chan_out_scaled
    ca_passb<<<dim3(BB * NH, NTOK / 128), 256, 0, stream>>>(
        qkvb, qkvsb, cob, colmax, out1);
}
